// DistanceLoss_41403484733685
// MI455X (gfx1250) — compile-verified, matrix operand order checked by reading, not run on hardware
//
#include <hip/hip_runtime.h>
#include <hip/hip_bf16.h>

typedef __attribute__((ext_vector_type(16))) _Float16 v16h;
typedef __attribute__((ext_vector_type(8)))  float    v8f;

#define B_ROWS        262144
#define C_DIM         512
#define NCLS          5
#define GROUP_ROWS    32
#define PAD_K         40   /* padded k-run per column: 40 halves = 80 B -> 2-way LDS conflicts */
#define NUM_WG        1024
#define GROUPS_TOTAL  (B_ROWS / GROUP_ROWS)     /* 8192 */
#define GROUPS_PER_WG (GROUPS_TOTAL / NUM_WG)   /* 8    */
#define THREADS       256

// ---------------------------------------------------------------------------
// Zero the workspace (5*512 f32 partial sums + 5 u32 counts, rounded up).
// ---------------------------------------------------------------------------
__global__ __launch_bounds__(256) void zero_ws_kernel(float* ws, int n) {
  int i = blockIdx.x * blockDim.x + threadIdx.x;
  if (i < n) ws[i] = 0.0f;
}

// ---------------------------------------------------------------------------
// Main pass: streaming softmax + one-hot-GEMM segmented sum via WMMA.
//   D[m][n] = sum_k onehot(target[k]==m) * softmax(x)[k][n]
// Each WG: 256 rows (8 groups of 32). Each wave: 4 rows of softmax per group,
// then 4 column tiles (64 cols) of v_wmma_f32_16x16x32_f16 accumulation.
// ---------------------------------------------------------------------------
__global__ __launch_bounds__(THREADS) void center_accum_kernel(
    const float* __restrict__ inp,     // [B_ROWS, C_DIM]
    const int*   __restrict__ target,  // [B_ROWS]
    float*       __restrict__ partial, // [NCLS, C_DIM] global accumulators
    unsigned*    __restrict__ gcnt) {  // [NCLS] class counts
  __shared__ _Float16 probsT[C_DIM * PAD_K];  // [col][k(padded)], 40 KB
  __shared__ int      tgt[GROUP_ROWS];
  __shared__ unsigned cnt[8];

  const int tid  = threadIdx.x;
  const int lane = tid & 31;
  const int wid  = tid >> 5;
  if (tid < 8) cnt[tid] = 0u;

  v8f acc[4] = {};  // persistent 16(class) x 16(col) f32 tiles for 4 col tiles

  const int g0 = blockIdx.x * GROUPS_PER_WG;
  for (int g = 0; g < GROUPS_PER_WG; ++g) {
    const int rowbase = (g0 + g) * GROUP_ROWS;
    __syncthreads();  // probsT/tgt reuse fence vs previous group's WMMA reads

    // ---- Phase A: softmax of 4 rows per wave, f16 probs -> LDS (transposed,
    // padded). Lane owns columns {lane + 32*i}: each global load is one
    // contiguous 128 B line per wave; NT hint keeps the 512 MB single-use
    // stream out of L2. LDS write stride/lane = 80 B -> max 2-way conflict.
    #pragma unroll
    for (int j = 0; j < 4; ++j) {
      const int k = wid * 4 + j;          // row-in-group 0..31
      const int r = rowbase + k;
      const float* rowp = inp + (size_t)r * C_DIM + lane;
      float x[16];
      #pragma unroll
      for (int i = 0; i < 16; ++i)
        x[i] = __builtin_nontemporal_load(rowp + 32 * i);
      float m = x[0];
      #pragma unroll
      for (int i = 1; i < 16; ++i) m = fmaxf(m, x[i]);
      #pragma unroll
      for (int off = 1; off < 32; off <<= 1) m = fmaxf(m, __shfl_xor(m, off, 32));
      float s = 0.0f;
      #pragma unroll
      for (int i = 0; i < 16; ++i) { x[i] = __expf(x[i] - m); s += x[i]; }
      #pragma unroll
      for (int off = 1; off < 32; off <<= 1) s += __shfl_xor(s, off, 32);
      const float inv = 1.0f / s;
      #pragma unroll
      for (int i = 0; i < 16; ++i) {
        const int col = lane + 32 * i;
        probsT[col * PAD_K + k] = (_Float16)(x[i] * inv);
      }
      if (lane == 0) {
        const int t = target[r];
        tgt[k] = t;
        atomicAdd(&cnt[t], 1u);
      }
    }
    __syncthreads();

    // ---- Phase B: one-hot A fragment (ISA 16-bit A layout: lanes 0-15 hold
    // K in {0..7, 16..23}; lanes 16-31 hold K in {8..15, 24..31})
    const int mrow  = lane & 15;
    const int kbase = (lane >> 4) * 8;
    v16h afrag;
    #pragma unroll
    for (int h = 0; h < 16; ++h) {
      const int k = kbase + ((h < 8) ? h : (h + 8));
      afrag[h] = (tgt[k] == mrow) ? (_Float16)1.0f : (_Float16)0.0f;
    }

    #pragma unroll
    for (int t = 0; t < 4; ++t) {
      const int col = (wid * 4 + t) * 16 + mrow;  // B fragment: lane -> column
      const _Float16* pb = &probsT[col * PAD_K + kbase];  // 16B-aligned runs
      v16h bfrag;
      #pragma unroll
      for (int h = 0; h < 8; ++h) {  // two contiguous 8-half runs (K, K+16)
        bfrag[h]     = pb[h];
        bfrag[h + 8] = pb[h + 16];
      }
      acc[t] = __builtin_amdgcn_wmma_f32_16x16x32_f16(
          /*neg_a=*/false, afrag, /*neg_b=*/false, bfrag,
          /*c_mod=*/(short)0, acc[t], /*reuse_a=*/false, /*reuse_b=*/false);
    }
  }

  // ---- Flush: C/D layout (§7.12.2): VGPR j, lane l -> M = j + 8*(l>=16),
  // N = l&15. Only classes 0..4 are live.
  #pragma unroll
  for (int t = 0; t < 4; ++t) {
    const int colb = (wid * 4 + t) * 16 + (lane & 15);
    #pragma unroll
    for (int j = 0; j < 8; ++j) {
      const int mm = j + ((lane >= 16) ? 8 : 0);
      if (mm < NCLS) atomicAdd(&partial[mm * C_DIM + colb], acc[t][j]);
    }
  }
  __syncthreads();
  if (tid < NCLS) atomicAdd(&gcnt[tid], cnt[tid]);
}

// ---------------------------------------------------------------------------
// Finalize: centers = partial/max(count,1); two 512-wide MSEs; scalar loss.
// ---------------------------------------------------------------------------
__global__ __launch_bounds__(512) void finalize_kernel(
    const float* __restrict__ partial, const unsigned* __restrict__ gcnt,
    float* __restrict__ out) {
  __shared__ float red01[16];
  __shared__ float red23[16];
  const int c    = threadIdx.x;   // one column per thread
  const int lane = c & 31;
  const int wid  = c >> 5;

  const float n0 = fmaxf((float)gcnt[0], 1.0f);
  const float n1 = fmaxf((float)gcnt[1], 1.0f);
  const float n2 = fmaxf((float)gcnt[2], 1.0f);
  const float n3 = fmaxf((float)gcnt[3], 1.0f);

  const float c0 = partial[0 * C_DIM + c] / n0;
  const float c1 = partial[1 * C_DIM + c] / n1;
  const float c2 = partial[2 * C_DIM + c] / n2;
  const float c3 = partial[3 * C_DIM + c] / n3;

  float d01 = (c0 - c1) * (c0 - c1);
  float d23 = (c2 - c3) * (c2 - c3);
  #pragma unroll
  for (int off = 1; off < 32; off <<= 1) {
    d01 += __shfl_xor(d01, off, 32);
    d23 += __shfl_xor(d23, off, 32);
  }
  if (lane == 0) { red01[wid] = d01; red23[wid] = d23; }
  __syncthreads();
  if (c == 0) {
    float s01 = 0.0f, s23 = 0.0f;
    #pragma unroll
    for (int i = 0; i < 16; ++i) { s01 += red01[i]; s23 += red23[i]; }
    const float mse01 = s01 / (float)C_DIM;
    const float mse23 = s23 / (float)C_DIM;
    out[0] = 1e-8f + (1.0f - mse01) + (1.0f - mse23);
  }
}

// ---------------------------------------------------------------------------
extern "C" void kernel_launch(void* const* d_in, const int* in_sizes, int n_in,
                              void* d_out, int out_size, void* d_ws, size_t ws_size,
                              hipStream_t stream) {
  const float* inp    = (const float*)d_in[0];
  const int*   target = (const int*)d_in[1];   // per harness: integer -> const int*
  float*       out    = (float*)d_out;

  float*    partial = (float*)d_ws;                         // [5*512] f32
  unsigned* gcnt    = (unsigned*)(partial + NCLS * C_DIM);  // [5] u32

  const int zero_n = NCLS * C_DIM + 8;  // partials + counts (0.0f bits == 0u)
  zero_ws_kernel<<<(zero_n + 255) / 256, 256, 0, stream>>>(partial, zero_n);

  center_accum_kernel<<<NUM_WG, THREADS, 0, stream>>>(inp, target, partial, gcnt);

  finalize_kernel<<<1, 512, 0, stream>>>(partial, gcnt, out);
}
